// SupConLossTopK_29497835389893
// MI455X (gfx1250) — compile-verified, matrix operand order checked by reading, not run on hardware
//
#include <hip/hip_runtime.h>
#include <hip/hip_bf16.h>
#include <math.h>

// ---------------------------------------------------------------------------
// SupCon loss with top-k hard positives + random negatives, fused on CDNA5.
// K = 8192 anchors, D = 256 dims, labels in [0,64).
//   sim = (f/|f|)(f/|f|)^T / 0.1  -- f16 WMMA tiles, never materialized.
// Each block (1 wave) owns 32 anchor rows = two 16-row WMMA tiles that share
// every B fragment (16 WMMAs per 16-column tile, 16 B loads). B K-chunks are
// software-pipelined 2-deep so next loads issue before current WMMAs.
// Streaming per-row state: row max, exact top-6 positives, reservoir-30 negs.
// ---------------------------------------------------------------------------

typedef __attribute__((ext_vector_type(16))) _Float16 v16h;
typedef __attribute__((ext_vector_type(8)))  _Float16 v8h;
typedef __attribute__((ext_vector_type(8)))  float    v8f;

#define KROWS 8192
#define DDIM  256
#define TEMP_INV 10.0f   // 1 / TEMPERATURE
#define MAX_POS 6
#define MAX_NEG 30
#define ROWS_PER_BLOCK 32
#define NTILES (KROWS / 16)

// ---------------------------------------------------------------------------
// Kernel 1: L2-normalize each row of features (f32) -> f16 matrix in ws.
// ---------------------------------------------------------------------------
__global__ __launch_bounds__(32) void normalize_f16_kernel(
    const float* __restrict__ feat, _Float16* __restrict__ out)
{
    const int row  = blockIdx.x;
    const int lane = threadIdx.x;
    const float* r = feat + (size_t)row * DDIM;

    float4 a = *(const float4*)(r + lane * 8);
    float4 b = *(const float4*)(r + lane * 8 + 4);
    float ss = a.x*a.x + a.y*a.y + a.z*a.z + a.w*a.w
             + b.x*b.x + b.y*b.y + b.z*b.z + b.w*b.w;
    #pragma unroll
    for (int m = 16; m >= 1; m >>= 1) ss += __shfl_xor(ss, m, 32);

    const float inv = 1.0f / fmaxf(sqrtf(ss), 1e-12f);

    v8h o;
    o[0] = (_Float16)(a.x * inv); o[1] = (_Float16)(a.y * inv);
    o[2] = (_Float16)(a.z * inv); o[3] = (_Float16)(a.w * inv);
    o[4] = (_Float16)(b.x * inv); o[5] = (_Float16)(b.y * inv);
    o[6] = (_Float16)(b.z * inv); o[7] = (_Float16)(b.w * inv);
    *(v8h*)(out + (size_t)row * DDIM + lane * 8) = o;
}

__global__ void zero_out_kernel(float* out) { out[0] = 0.0f; }

// Hash PRNG for reservoir sampling (stands in for jax.random key 42)
__device__ __forceinline__ unsigned hash_u32(unsigned x)
{
    x ^= 42u;
    x *= 2654435761u; x ^= x >> 16;
    x *= 2246822519u; x ^= x >> 13;
    x *= 3266489917u; x ^= x >> 16;
    return x;
}

// 16x32 f16 fragment (A or B^T) at K-chunk kc.
// Layout (CDNA5 ISA §7.12.2): lane covers K = kc*32 + {hi*8..+7, 16+hi*8..+7}
__device__ __forceinline__ v16h load_frag(const _Float16* __restrict__ rowp,
                                          int kc, int hi)
{
    const _Float16* p = rowp + kc * 32 + hi * 8;
    v8h lo  = *(const v8h*)(p);
    v8h hi8 = *(const v8h*)(p + 16);
    v16h f;
    #pragma unroll
    for (int t = 0; t < 8; ++t) { f[t] = lo[t]; f[t + 8] = hi8[t]; }
    return f;
}

// ---------------------------------------------------------------------------
// Kernel 3: fused sim-GEMM (f16 WMMA) + streaming SupCon reduction.
// ---------------------------------------------------------------------------
__global__ __launch_bounds__(32) void supcon_wmma_kernel(
    const _Float16* __restrict__ fn,   // [KROWS, DDIM] normalized f16
    const int* __restrict__ labels,    // [KROWS]
    float* __restrict__ out)           // scalar accumulator
{
    __shared__ int   lds_lab[KROWS];                 // 32 KB
    __shared__ float tile[ROWS_PER_BLOCK][17];       // 32x16 sim tile (+pad)
    __shared__ float posl[ROWS_PER_BLOCK][MAX_POS];  // top-6 positive sims
    __shared__ float negl[ROWS_PER_BLOCK][MAX_NEG];  // negative reservoir

    const int lane = threadIdx.x;      // wave32
    const int m    = lane & 15;
    const int hi   = lane >> 4;
    const int row0 = blockIdx.x * ROWS_PER_BLOCK;

    // preload all labels into LDS (broadcast reads later)
    for (int i = lane; i < KROWS; i += 32) lds_lab[i] = labels[i];

    // A fragments for the two 16-row tiles (resident in VGPRs for the sweep)
    const _Float16* arow0 = fn + (size_t)(row0 + m) * DDIM;
    const _Float16* arow1 = fn + (size_t)(row0 + 16 + m) * DDIM;
    v16h afrag0[8], afrag1[8];
    #pragma unroll
    for (int kc = 0; kc < 8; ++kc) afrag0[kc] = load_frag(arow0, kc, hi);
    #pragma unroll
    for (int kc = 0; kc < 8; ++kc) afrag1[kc] = load_frag(arow1, kc, hi);

    // per-row streaming state: each lane owns one of the 32 rows
    #pragma unroll
    for (int j = 0; j < MAX_POS; ++j) posl[lane][j] = -3.0e38f;
    int   n_pos = 0, n_neg = 0;
    float rowmax = -3.0e38f;
    const int grow  = row0 + lane;
    const int mylab = labels[grow];
    __syncthreads();

    for (int ct = 0; ct < NTILES; ++ct) {
        const int col0 = ct * 16;

        // prefetch next column tile's rows into the WGP cache
        if (ct + 1 < NTILES)
            __builtin_prefetch(fn + (size_t)(col0 + 16 + m) * DDIM, 0, 2);

        const _Float16* brow = fn + (size_t)(col0 + m) * DDIM;

        // 2-deep software pipeline over the 8 K-chunks: issue chunk kc+1's
        // loads BEFORE chunk kc's WMMA pair so load latency overlaps math.
        v16h cur = load_frag(brow, 0, hi);
        v8f acc0 = {}, acc1 = {};
        #pragma unroll
        for (int kc = 0; kc < 8; ++kc) {
            v16h nxt;
            if (kc < 7) nxt = load_frag(brow, kc + 1, hi);
            acc0 = __builtin_amdgcn_wmma_f32_16x16x32_f16(
                       false, afrag0[kc], false, cur,
                       (short)0, acc0, false, false);
            acc1 = __builtin_amdgcn_wmma_f32_16x16x32_f16(
                       false, afrag1[kc], false, cur,
                       (short)0, acc1, false, false);
            cur = nxt;
        }

        // C layout: lane holds column N=lane&15; VGPR v holds row M=v+hi*8
        #pragma unroll
        for (int v = 0; v < 8; ++v) {
            tile[hi * 8 + v][m]      = acc0[v];
            tile[16 + hi * 8 + v][m] = acc1[v];
        }
        __syncthreads();

        // scalar phase: all 32 lanes, one row each, 16 columns
        const int r = lane;
        #pragma unroll
        for (int j4 = 0; j4 < 4; ++j4) {
            const int4 lab4 = *(const int4*)&lds_lab[col0 + j4 * 4];
            #pragma unroll
            for (int t = 0; t < 4; ++t) {
                const int j   = j4 * 4 + t;
                const int lab = (t == 0) ? lab4.x : (t == 1) ? lab4.y
                              : (t == 2) ? lab4.z : lab4.w;
                const float s = tile[r][j] * TEMP_INV;
                rowmax = fmaxf(rowmax, s);           // row max incl diagonal
                const int col = col0 + j;
                if (col == grow) continue;           // skip diagonal
                if (lab == mylab) {
                    ++n_pos;                         // exact top-6 insert
                    if (s > posl[r][MAX_POS - 1]) {
                        int p = MAX_POS - 1;
                        while (p > 0 && posl[r][p - 1] < s) {
                            posl[r][p] = posl[r][p - 1]; --p;
                        }
                        posl[r][p] = s;
                    }
                } else {
                    ++n_neg;                         // reservoir sample of 30
                    if (n_neg <= MAX_NEG) {
                        negl[r][n_neg - 1] = s;
                    } else {
                        const unsigned h = hash_u32((unsigned)grow * 8192u
                                                    + (unsigned)col);
                        // uniform in [0, n_neg) without integer division
                        const unsigned idx = __umulhi(h, (unsigned)n_neg);
                        if (idx < MAX_NEG) negl[r][idx] = s;
                    }
                }
            }
        }
        __syncthreads();
    }

    // finalize per-row loss (POS_NEG_RATIO = 1.0 -> int(n_neg*1.0) == n_neg)
    {
        const int r = lane;
        int mp = min(n_pos, max(1, min(MAX_POS, n_neg)));
        int mn = min(n_neg, MAX_NEG);
        const bool valid = (n_pos > 0) && (n_neg > 0);
        if (!valid) { mp = 0; mn = 0; }

        float num = 0.0f;
        for (int j = 0; j < mp; ++j) num += __expf(posl[r][j] - rowmax);
        float den = num;
        for (int j = 0; j < mn; ++j) den += __expf(negl[r][j] - rowmax);

        const float ratio = (den > 0.0f) ? (num / den) : 0.0f;
        const float loss  = -logf(fmaxf(ratio, 1e-8f));
        atomicAdd(out, loss * (1.0f / (float)KROWS));
    }
}

// ---------------------------------------------------------------------------
// Host-side launch
// ---------------------------------------------------------------------------
extern "C" void kernel_launch(void* const* d_in, const int* in_sizes, int n_in,
                              void* d_out, int out_size, void* d_ws, size_t ws_size,
                              hipStream_t stream)
{
    const float* features = (const float*)d_in[0];   // [8192, 256] f32
    const int*   labels   = (const int*)d_in[1];     // [8192]
    float*       out      = (float*)d_out;           // scalar mean loss

    _Float16* fnorm = (_Float16*)d_ws;               // 4 MB scratch

    normalize_f16_kernel<<<KROWS, 32, 0, stream>>>(features, fnorm);
    zero_out_kernel<<<1, 1, 0, stream>>>(out);
    supcon_wmma_kernel<<<KROWS / ROWS_PER_BLOCK, 32, 0, stream>>>(fnorm, labels, out);
}
